// Gate_30485677867853
// MI455X (gfx1250) — matrix operand order repaired and checked by reading, not compiled, not run on hardware
//
#include <hip/hip_runtime.h>

// ---------------------------------------------------------------------------
// MoE sigmoid router for MI455X (gfx1250, wave32, WMMA).
//  scores = sigmoid(x @ W^T); group-limited top-k routing.
//  x: [T,7168] f32 (streamed from HBM once -> bandwidth bound, ~20us floor)
//  W: [256,7168] f32 (pre-converted to bf16 in d_ws, L2 resident)
//  out: weights [T,8] f32  followed by  indices [T,8] i32
// ---------------------------------------------------------------------------

typedef __attribute__((ext_vector_type(16))) __bf16 bf16x16;
typedef __attribute__((ext_vector_type(8)))  __bf16 bf16x8;
typedef __attribute__((ext_vector_type(8)))  float  f32x8;

#define DDIM   7168          // hidden dim
#define NEXP   256           // experts
#define NGRP   8             // groups
#define GSZ    32            // experts per group
#define TOPG   4             // top groups
#define TOPK   8             // top experts
#define MTILE  32            // tokens per workgroup
#define KC     256           // K chunk staged in LDS
#define AS     264           // LDS A row stride (bf16 elems, padded)
#define SC     266           // LDS score row stride (f32 elems, padded)

// fp32 -> bf16, round to nearest even (pair packed into one dword)
__device__ __forceinline__ unsigned f2bf2(float lo, float hi) {
  unsigned ul = __float_as_uint(lo);
  unsigned uh = __float_as_uint(hi);
  ul = (ul + 0x7FFFu + ((ul >> 16) & 1u)) >> 16;
  uh = (uh + 0x7FFFu + ((uh >> 16) & 1u)) & 0xFFFF0000u;
  return ul | uh;
}

__device__ __forceinline__ f32x8 wmma_bf16(bf16x16 a, bf16x16 b, f32x8 c) {
  // D = A(16x32 bf16) x B(32x16 bf16) + C(16x16 f32)
  return __builtin_amdgcn_wmma_f32_16x16x32_bf16(
      /*neg_a=*/false, a, /*neg_b=*/false, b,
      /*c_mod=*/(short)0, c, /*reuse_a=*/false, /*reuse_b=*/false);
}

#define CAT16(lo, hi) \
  __builtin_shufflevector(lo, hi, 0,1,2,3,4,5,6,7,8,9,10,11,12,13,14,15)

// ---------------------------------------------------------------------------
// Kernel 1: convert weight fp32 -> bf16 into scratch (once per launch).
// 8 elements / thread.
// ---------------------------------------------------------------------------
__global__ __launch_bounds__(256) void wconv_kernel(
    const float* __restrict__ w, unsigned* __restrict__ o) {
  const int i = blockIdx.x * blockDim.x + threadIdx.x;   // 4-dword packet id
  const float4* p = (const float4*)w + (size_t)i * 2;
  float4 a = p[0], b = p[1];
  uint4 r;
  r.x = f2bf2(a.x, a.y);
  r.y = f2bf2(a.z, a.w);
  r.z = f2bf2(b.x, b.y);
  r.w = f2bf2(b.z, b.w);
  ((uint4*)o)[i] = r;
}

// ---------------------------------------------------------------------------
// Kernel 2: GEMM (bf16 WMMA) + sigmoid + group-limited top-k routing.
// Grid: T/32 blocks x 256 threads (8 waves). Wave w owns experts [32w,32w+32).
// ---------------------------------------------------------------------------
__global__ __launch_bounds__(256) void gate_kernel(
    const float* __restrict__ x,
    const unsigned short* __restrict__ wbf,  // bf16 weights [NEXP][DDIM]
    float* __restrict__ ow,                  // [T][8] routing weights
    int* __restrict__ oi) {                  // [T][8] expert indices
  __shared__ __attribute__((aligned(16))) unsigned short sA[MTILE * AS];
  __shared__ __attribute__((aligned(16))) float sS[MTILE * SC];
  __shared__ float sG[MTILE * NGRP];

  const int tid   = threadIdx.x;
  const int lane  = tid & 31;
  const int wave  = tid >> 5;
  const int wbase = wave * 32;               // first expert of this wave
  const int t0    = blockIdx.x * MTILE;      // first token of this block

  // --- cooperative x staging coordinates (32 rows x 8 threads x 32 floats)
  const int sr = tid >> 3;
  const int sc = (tid & 7) * 32;
  const float* xrow = x + (size_t)(t0 + sr) * DDIM + sc;

  // --- WMMA fragment coordinates (contiguous K order k = 16*half + s, same for A and B)
  const int arow  = lane & 15;               // M row / N column / expert lane
  const int ahalf = lane >> 4;               // half-wave K split
  const int kA    = ahalf * 16;              // A: lanes 16-31 hold K=16..31
  const int kB    = ahalf * 16;              // B: lanes 16-31 hold K=16..31

  f32x8 acc[2][2] = {};                      // [m-tile][n-tile]

  for (int kc = 0; kc < DDIM; kc += KC) {
    __syncthreads();                         // previous chunk fully consumed
    // ---- stage x chunk: fp32 global -> bf16 LDS (RNE) ----
    {
      const float4* p = (const float4*)(xrow + kc);
      float4 fv[8];
#pragma unroll
      for (int i = 0; i < 8; ++i) fv[i] = p[i];
      uint4 q[4];
#pragma unroll
      for (int i = 0; i < 4; ++i) {
        q[i].x = f2bf2(fv[2 * i].x, fv[2 * i].y);
        q[i].y = f2bf2(fv[2 * i].z, fv[2 * i].w);
        q[i].z = f2bf2(fv[2 * i + 1].x, fv[2 * i + 1].y);
        q[i].w = f2bf2(fv[2 * i + 1].z, fv[2 * i + 1].w);
      }
      uint4* dst = (uint4*)(&sA[sr * AS + sc]);
#pragma unroll
      for (int i = 0; i < 4; ++i) dst[i] = q[i];
      if (kc + KC < DDIM) __builtin_prefetch(xrow + kc + KC, 0, 1);
    }
    __syncthreads();

    // ---- 8 K-steps of 32, 4 WMMAs each ----
#pragma unroll
    for (int s = 0; s < KC / 32; ++s) {
      const int k0 = s * 32;
      // A fragments for the two 16-token M tiles (from LDS)
      bf16x8 a0lo = *(const bf16x8*)&sA[arow * AS + k0 + kA];
      bf16x8 a0hi = *(const bf16x8*)&sA[arow * AS + k0 + kA + 8];
      bf16x8 a1lo = *(const bf16x8*)&sA[(16 + arow) * AS + k0 + kA];
      bf16x8 a1hi = *(const bf16x8*)&sA[(16 + arow) * AS + k0 + kA + 8];
      bf16x16 a0 = CAT16(a0lo, a0hi);
      bf16x16 a1 = CAT16(a1lo, a1hi);
      // B fragments for the wave's two 16-expert N tiles (L2-resident bf16)
      const unsigned short* bp =
          wbf + (size_t)(wbase + arow) * DDIM + kc + k0 + kB;
      bf16x16 b0 = *(const bf16x16*)bp;
      bf16x16 b1 = *(const bf16x16*)(bp + (size_t)16 * DDIM);
      acc[0][0] = wmma_bf16(a0, b0, acc[0][0]);
      acc[0][1] = wmma_bf16(a0, b1, acc[0][1]);
      acc[1][0] = wmma_bf16(a1, b0, acc[1][0]);
      acc[1][1] = wmma_bf16(a1, b1, acc[1][1]);
    }
  }

  // ---- epilogue: sigmoid, scatter scores to LDS score board ----
#pragma unroll
  for (int mt = 0; mt < 2; ++mt)
#pragma unroll
    for (int nt = 0; nt < 2; ++nt)
#pragma unroll
      for (int i = 0; i < 8; ++i) {
        const int m = mt * 16 + ahalf * 8 + i;        // C layout: VGPR i = row
        const int e = wbase + nt * 16 + arow;
        const float v = acc[mt][nt][i];
        sS[m * SC + e] = 1.0f / (1.0f + __expf(-v));
      }
  __syncthreads();

  // ---- group maxes: 256 threads = 32 tokens x 8 groups ----
  {
    const int t = tid >> 3, g = tid & 7;
    const float* row = &sS[t * SC + g * GSZ];
    float m = row[0];
#pragma unroll
    for (int j = 1; j < GSZ; ++j) m = fmaxf(m, row[j]);
    sG[t * NGRP + g] = m;
  }
  __syncthreads();

  // ---- per-token routing: lanes 0..31 of wave 0, one token each ----
  if (tid < MTILE) {
    const int t = tid;
    float gm[NGRP];
#pragma unroll
    for (int g = 0; g < NGRP; ++g) gm[g] = sG[t * NGRP + g];
    int gsel[TOPG];
#pragma unroll
    for (int i = 0; i < TOPG; ++i) {        // top-4 groups (scores in (0,1))
      int bg = 0;
      float bv = gm[0];
#pragma unroll
      for (int g = 1; g < NGRP; ++g)
        if (gm[g] > bv) { bv = gm[g]; bg = g; }
      gsel[i] = bg;
      gm[bg] = -1.0f;
    }
    float wv[TOPK];
    int   wi[TOPK];
    float sum = 0.0f;
    for (int k = 0; k < TOPK; ++k) {        // 8x argmax over 4x32 kept scores
      float bv = -1.0f;
      int bi = 0;
#pragma unroll
      for (int i = 0; i < TOPG; ++i) {
        const int gb = gsel[i] * GSZ;
        const int base = t * SC + gb;
        for (int j = 0; j < GSZ; ++j) {
          const float v = sS[base + j];
          if (v > bv) { bv = v; bi = gb + j; }
        }
      }
      sS[t * SC + bi] = -1.0f;              // knock out selected expert
      wv[k] = bv;
      wi[k] = bi;
      sum += bv;
    }
    const float scale = 2.5f / sum;         // ROUTE_SCALE / sum(selected)
    const int tok = t0 + t;
#pragma unroll
    for (int k = 0; k < TOPK; ++k) {
      ow[tok * TOPK + k] = wv[k] * scale;
      oi[tok * TOPK + k] = wi[k];
    }
  }
}

// ---------------------------------------------------------------------------
extern "C" void kernel_launch(void* const* d_in, const int* in_sizes, int n_in,
                              void* d_out, int out_size, void* d_ws,
                              size_t ws_size, hipStream_t stream) {
  const float* x = (const float*)d_in[0];   // [T, 7168] f32
  const float* w = (const float*)d_in[1];   // [256, 7168] f32
  const int T = in_sizes[0] / DDIM;         // 16384

  // scratch: bf16 weights, NEXP*DDIM*2 = 3.67 MB (ws must cover this)
  unsigned short* wbf = (unsigned short*)d_ws;

  // 1) convert weights fp32 -> bf16 (8 elems / thread)
  {
    const int nthreads = (NEXP * DDIM) / 8;
    wconv_kernel<<<nthreads / 256, 256, 0, stream>>>(w, (unsigned*)wbf);
  }

  // 2) fused gemm + sigmoid + grouped top-k routing
  {
    float* ow = (float*)d_out;              // [T,8] weights
    int* oi = (int*)d_out + (size_t)T * TOPK;  // [T,8] indices (i32)
    gate_kernel<<<T / MTILE, 256, 0, stream>>>(x, wbf, ow, oi);
  }
}
